// GNN_83356725281053
// MI455X (gfx1250) — matrix operand order repaired and checked by reading, not compiled, not run on hardware
//
#include <hip/hip_runtime.h>
#include <hip/hip_bf16.h>

#define N_NODES  50000
#define N_EDGES  800000
#define D        128
#define N_GRAPHS 256
#define N_OUT    10
#define NLAYERS  5
#define SCAN_BLOCKS 196   // ceil(N_NODES/256)

typedef __attribute__((ext_vector_type(16))) __bf16 v16bf;
typedef __attribute__((ext_vector_type(8)))  float  v8f;

struct U4 { unsigned int x, y, z, w; };
struct U2 { unsigned int x, y; };
union BFrag { v16bf v; unsigned short s[16]; U4 q[2]; };

__device__ __forceinline__ unsigned short f2bf(float f) {
  union { float f; unsigned int u; } c; c.f = f;
  unsigned int u = c.u;
  return (unsigned short)((u + 0x7FFFu + ((u >> 16) & 1u)) >> 16);  // RNE
}

__device__ __forceinline__ float bf2f(unsigned int s) {
  union { unsigned int u; float f; } c; c.u = s << 16;
  return c.f;
}

__device__ __forceinline__ void atomAddF(float* p, float v) {
  __hip_atomic_fetch_add(p, v, __ATOMIC_RELAXED, __HIP_MEMORY_SCOPE_AGENT);
}

// ---------------- utility ----------------
__global__ void k_fill(float* p, float v, int n) {
  int i = blockIdx.x * blockDim.x + threadIdx.x;
  if (i < n) p[i] = v;
}

// in-degree histogram (one-time int atomics)
__global__ void k_hist(const int* __restrict__ dst, unsigned int* __restrict__ cnt) {
  int i = blockIdx.x * blockDim.x + threadIdx.x;
  if (i < N_EDGES) atomicAdd(&cnt[dst[i]], 1u);
}

// dinv[i] = rsqrt(indeg + 1)  (self-loop included)
__global__ void k_dinv(const unsigned int* __restrict__ cnt, float* __restrict__ dinv) {
  int i = blockIdx.x * blockDim.x + threadIdx.x;
  if (i < N_NODES) dinv[i] = rsqrtf((float)cnt[i] + 1.0f);
}

// ---------------- two-level exclusive scan for rowptr ----------------
__global__ void k_scan1(const unsigned int* __restrict__ cnt, unsigned int* __restrict__ pre,
                        unsigned int* __restrict__ blk) {
  __shared__ unsigned int tmp[256];
  int i = blockIdx.x * 256 + threadIdx.x;
  unsigned int v = (i < N_NODES) ? cnt[i] : 0u;
  tmp[threadIdx.x] = v;
  __syncthreads();
  for (int off = 1; off < 256; off <<= 1) {
    unsigned int add = (threadIdx.x >= off) ? tmp[threadIdx.x - off] : 0u;
    __syncthreads();
    tmp[threadIdx.x] += add;
    __syncthreads();
  }
  pre[i] = tmp[threadIdx.x] - v;                 // exclusive within block
  if (threadIdx.x == 255) blk[blockIdx.x] = tmp[255];
}

__global__ void k_scan2(const unsigned int* __restrict__ blk, unsigned int* __restrict__ blkoff) {
  __shared__ unsigned int tmp[256];
  int t = threadIdx.x;
  unsigned int v = (t < SCAN_BLOCKS) ? blk[t] : 0u;
  tmp[t] = v;
  __syncthreads();
  for (int off = 1; off < 256; off <<= 1) {
    unsigned int add = (t >= off) ? tmp[t - off] : 0u;
    __syncthreads();
    tmp[t] += add;
    __syncthreads();
  }
  if (t < SCAN_BLOCKS) blkoff[t] = tmp[t] - v;   // exclusive across blocks
}

__global__ void k_scan3(const unsigned int* __restrict__ pre, const unsigned int* __restrict__ blkoff,
                        int* __restrict__ rowptr) {
  int i = blockIdx.x * 256 + threadIdx.x;
  if (i < N_NODES) rowptr[i] = (int)(pre[i] + blkoff[i >> 8]);
  if (i == N_NODES) rowptr[N_NODES] = N_EDGES;
}

// scatter edges into CSR slots; store src index and edge norm
__global__ void k_scatter(const int* __restrict__ src, const int* __restrict__ dst,
                          const float* __restrict__ dinv, const int* __restrict__ rowptr,
                          unsigned int* __restrict__ fill,
                          int* __restrict__ esrc, float* __restrict__ ew) {
  int e = blockIdx.x * blockDim.x + threadIdx.x;
  if (e >= N_EDGES) return;
  int s = src[e], d = dst[e];
  unsigned int off = atomicAdd(&fill[d], 1u);
  int pos = rowptr[d] + (int)off;
  esrc[pos] = s;
  ew[pos]   = dinv[s] * dinv[d];
}

// ---------------- bf16 staging ----------------
// x (f32) -> bf16 row-major, 8 elements/thread
__global__ void k_cvt(const float* __restrict__ x, unsigned short* __restrict__ hbf) {
  int i = blockIdx.x * blockDim.x + threadIdx.x;     // N_NODES*D/8 threads
  size_t base = (size_t)i * 8;
  float4 f0 = *(const float4*)(x + base);
  float4 f1 = *(const float4*)(x + base + 4);
  union { unsigned short s[8]; U4 q; } o;
  o.s[0] = f2bf(f0.x); o.s[1] = f2bf(f0.y); o.s[2] = f2bf(f0.z); o.s[3] = f2bf(f0.w);
  o.s[4] = f2bf(f1.x); o.s[5] = f2bf(f1.y); o.s[6] = f2bf(f1.z); o.s[7] = f2bf(f1.w);
  *(U4*)(hbf + base) = o.q;
}

// Swizzle the 5 weight matrices into WMMA bf16 B-fragment layout:
// frag (kb,nb): lane holds column nb*16+(lane&15), K = kb*32+(lane>=16?16:0)+j.
__global__ void k_prep_wbf(const float* __restrict__ W1, const float* __restrict__ Wm,
                           unsigned short* __restrict__ Wbf) {
  int tid = blockIdx.x * blockDim.x + threadIdx.x;
  if (tid >= NLAYERS * 1024) return;
  int l    = tid >> 10;
  int r    = tid & 1023;
  int frag = r >> 5;
  int lane = r & 31;
  int kb = frag >> 3, nb = frag & 7;
  const float* W = (l == 0) ? W1 : (Wm + (size_t)(l - 1) * D * D);
  int n     = nb * 16 + (lane & 15);
  int kbase = kb * 32 + ((lane >> 4) << 4);
  unsigned short* o = Wbf + (size_t)l * 16384 + (((size_t)frag * 32 + lane) << 4);
#pragma unroll
  for (int j = 0; j < 16; ++j) o[j] = f2bf(W[(size_t)(kbase + j) * D + n]);
}

// ---------------- WMMA GEMM: T[n,128] = A[n,128] @ W[128,128] ----------------
__global__ void k_gemm(const unsigned short* __restrict__ Abf,
                       const unsigned short* __restrict__ Wbf,
                       float* __restrict__ T, int ntiles) {
  __shared__ unsigned short ldsW[16384];   // 32 KB: 32 frags x 32 lanes x 16 bf16
  {
    const U4* s = (const U4*)Wbf;
    U4*       d = (U4*)ldsW;
    for (int i = threadIdx.x; i < 2048; i += 256) d[i] = s[i];
  }
  __syncthreads();

  int wave = threadIdx.x >> 5;
  int lane = threadIdx.x & 31;
  int mt   = blockIdx.x * 8 + wave;
  if (mt >= ntiles) return;

  int hi  = lane >> 4;
  int row = mt * 16 + (lane & 15);
  const unsigned short* arow = Abf + (size_t)row * D;

  v8f zero = {0.f, 0.f, 0.f, 0.f, 0.f, 0.f, 0.f, 0.f};
  v8f acc[8];
#pragma unroll
  for (int nb = 0; nb < 8; ++nb) acc[nb] = zero;

#pragma unroll
  for (int kb = 0; kb < 4; ++kb) {
    // A: lanes 0-15 hold K=0..15, lanes 16-31 K=16..31
    int bk = kb * 32 + hi * 16;
    const U4* ap = (const U4*)(arow + bk);
    BFrag a;
    a.q[0] = ap[0];
    a.q[1] = ap[1];
#pragma unroll
    for (int nb = 0; nb < 8; ++nb) {
      BFrag b;
      const U4* bp = (const U4*)(ldsW + (((kb * 8 + nb) * 32 + lane) << 4));
      b.q[0] = bp[0];
      b.q[1] = bp[1];
      acc[nb] = __builtin_amdgcn_wmma_f32_16x16x32_bf16(
          false, a.v, false, b.v, (short)0, acc[nb], false, false);
    }
  }

  int mbase = mt * 16 + hi * 8;
  int col0  = lane & 15;
#pragma unroll
  for (int v = 0; v < 8; ++v) {
    float* trow = T + (size_t)(mbase + v) * D + col0;
#pragma unroll
    for (int nb = 0; nb < 8; ++nb) trow[nb * 16] = acc[nb][v];
  }
}

// ---------------- fused atomic-free aggregation ----------------
// one wave per dst node: acc = dinv^2*t[node] + sum_e w_e*t[src_e];
// h = bf16(relu(acc + bias))
__global__ void k_agg(const int* __restrict__ rowptr, const int* __restrict__ esrc,
                      const float* __restrict__ ew, const float* __restrict__ t,
                      const float* __restrict__ dinv, const float* __restrict__ bias,
                      unsigned short* __restrict__ hbf) {
  int gtid = blockIdx.x * blockDim.x + threadIdx.x;
  int lane = gtid & 31, node = gtid >> 5;
  if (node >= N_NODES) return;

  float di = dinv[node];
  float sc = di * di;
  const float4* tself = (const float4*)(t + (size_t)node * D);
  float4 v = tself[lane];
  float ax = sc * v.x, ay = sc * v.y, az = sc * v.z, aw = sc * v.w;

  int beg = rowptr[node], end = rowptr[node + 1];
  for (int e = beg; e < end; ++e) {
    int s   = esrc[e];
    float w = ew[e];
    const float4* tp = (const float4*)(t + (size_t)s * D);
    float4 u = tp[lane];
    ax = fmaf(w, u.x, ax);
    ay = fmaf(w, u.y, ay);
    az = fmaf(w, u.z, az);
    aw = fmaf(w, u.w, aw);
  }

  const float4 b4 = *(const float4*)(bias + lane * 4);
  union { unsigned short s4[4]; U2 u; } o;
  o.s4[0] = f2bf(fmaxf(ax + b4.x, 0.0f));
  o.s4[1] = f2bf(fmaxf(ay + b4.y, 0.0f));
  o.s4[2] = f2bf(fmaxf(az + b4.z, 0.0f));
  o.s4[3] = f2bf(fmaxf(aw + b4.w, 0.0f));
  *(U2*)(hbf + (size_t)node * D + lane * 4) = o.u;
}

// ---------------- pooling + head ----------------
__global__ void k_pool(const unsigned short* __restrict__ hbf, const int* __restrict__ batch,
                       float* __restrict__ sums, float* __restrict__ cnts) {
  int gtid = blockIdx.x * blockDim.x + threadIdx.x;
  int lane = gtid & 31, node = gtid >> 5;
  if (node >= N_NODES) return;
  int b = batch[node];
  U2 p = *(const U2*)(hbf + (size_t)node * D + lane * 4);
  float* sp = sums + (size_t)b * D + lane * 4;
  atomAddF(sp + 0, bf2f(p.x & 0xFFFFu));
  atomAddF(sp + 1, bf2f(p.x >> 16));
  atomAddF(sp + 2, bf2f(p.y & 0xFFFFu));
  atomAddF(sp + 3, bf2f(p.y >> 16));
  if (lane == 0) atomAddF(&cnts[b], 1.0f);
}

__global__ void k_head(const float* __restrict__ sums, const float* __restrict__ cnts,
                       const float* __restrict__ fcW, const float* __restrict__ fcb,
                       float* __restrict__ out) {
  int g = threadIdx.x;                         // 256 graphs, 1 block
  float inv = 1.0f / fmaxf(cnts[g], 1.0f);
  float logits[N_OUT];
#pragma unroll
  for (int o = 0; o < N_OUT; ++o) logits[o] = fcb[o];
  for (int c = 0; c < D; ++c) {
    float pv = sums[(size_t)g * D + c] * inv;
#pragma unroll
    for (int o = 0; o < N_OUT; ++o) logits[o] += pv * fcW[c * N_OUT + o];
  }
  float m = logits[0];
#pragma unroll
  for (int o = 1; o < N_OUT; ++o) m = fmaxf(m, logits[o]);
  float se = 0.0f;
#pragma unroll
  for (int o = 0; o < N_OUT; ++o) se += __expf(logits[o] - m);
  float lse = m + __logf(se);
#pragma unroll
  for (int o = 0; o < N_OUT; ++o) out[(size_t)g * N_OUT + o] = logits[o] - lse;
}

// ---------------- host orchestration ----------------
extern "C" void kernel_launch(void* const* d_in, const int* in_sizes, int n_in,
                              void* d_out, int out_size, void* d_ws, size_t ws_size,
                              hipStream_t stream) {
  const float* x   = (const float*)d_in[0];
  const float* W1  = (const float*)d_in[1];
  const float* b1  = (const float*)d_in[2];
  const float* Wm  = (const float*)d_in[3];
  const float* bm  = (const float*)d_in[4];
  const float* fcW = (const float*)d_in[5];
  const float* fcb = (const float*)d_in[6];
  const int*   el  = (const int*)d_in[7];
  const int*   bat = (const int*)d_in[8];
  float*       out = (float*)d_out;

  const int* src = el;
  const int* dst = el + N_EDGES;

  // workspace carve-up (32-bit units, offsets stay 256B-aligned)
  float* ws = (float*)d_ws;
  size_t o = 0;
  float*        dinv   = ws + o;                   o += 50048;
  unsigned int* cnt    = (unsigned int*)(ws + o);  o += 50048;
  unsigned int* pre    = (unsigned int*)(ws + o);  o += 50176;   // SCAN_BLOCKS*256
  unsigned int* blk    = (unsigned int*)(ws + o);  o += 256;
  unsigned int* blkoff = (unsigned int*)(ws + o);  o += 256;
  int*          rowptr = (int*)(ws + o);           o += 50176;   // N_NODES+1 used
  unsigned int* fill   = (unsigned int*)(ws + o);  o += 50048;
  int*          esrc   = (int*)(ws + o);           o += N_EDGES;
  float*        ew     = ws + o;                   o += N_EDGES;
  float*        t      = ws + o;                   o += (size_t)N_NODES * D;
  unsigned short* hbf  = (unsigned short*)(ws + o); o += (size_t)N_NODES * D / 2;
  float*        sums   = ws + o;                   o += (size_t)N_GRAPHS * D;
  float*        cnts   = ws + o;                   o += 256;
  unsigned short* Wbf  = (unsigned short*)(ws + o);              // 5 * 16384 bf16

  // ---- build normalization + CSR (once per call) ----
  k_fill<<<SCAN_BLOCKS, 256, 0, stream>>>((float*)cnt, 0.0f, N_NODES);
  k_hist<<<N_EDGES / 256, 256, 0, stream>>>(dst, cnt);
  k_dinv<<<SCAN_BLOCKS, 256, 0, stream>>>(cnt, dinv);
  k_scan1<<<SCAN_BLOCKS, 256, 0, stream>>>(cnt, pre, blk);
  k_scan2<<<1, 256, 0, stream>>>(blk, blkoff);
  k_scan3<<<SCAN_BLOCKS + 1, 256, 0, stream>>>(pre, blkoff, rowptr);
  k_fill<<<SCAN_BLOCKS, 256, 0, stream>>>((float*)fill, 0.0f, N_NODES);
  k_scatter<<<N_EDGES / 256, 256, 0, stream>>>(src, dst, dinv, rowptr, fill, esrc, ew);

  // ---- stage weights (swizzled bf16) and input (bf16) ----
  k_prep_wbf<<<(NLAYERS * 1024 + 255) / 256, 256, 0, stream>>>(W1, Wm, Wbf);
  k_cvt<<<(N_NODES * D / 8) / 256, 256, 0, stream>>>(x, hbf);

  const int ntiles = N_NODES / 16;          // 3125
  const int gemm_blocks = (ntiles + 7) / 8; // 391

  for (int l = 0; l < NLAYERS; ++l) {
    k_gemm<<<gemm_blocks, 256, 0, stream>>>(hbf, Wbf + (size_t)l * 16384, t, ntiles);
    const float* bias = (l == 0) ? b1 : (bm + (size_t)(l - 1) * D);
    k_agg<<<(N_NODES * 32) / 256 + 1, 256, 0, stream>>>(rowptr, esrc, ew, t, dinv, bias, hbf);
  }

  // ---- mean pool + FC + log_softmax ----
  k_fill<<<(N_GRAPHS * D + 256 + 255) / 256, 256, 0, stream>>>(sums, 0.0f, N_GRAPHS * D + 256);
  k_pool<<<(N_NODES * 32) / 256, 256, 0, stream>>>(hbf, bat, sums, cnts);
  k_head<<<1, 256, 0, stream>>>(sums, cnts, fcW, fcb, out);
}
